// EdgeAwareTemporalGNN_76836964926274
// MI455X (gfx1250) — matrix operand order repaired and checked by reading, not compiled, not run on hardware
//
#include <hip/hip_runtime.h>
#include <math.h>

#define NNODES 10000
#define NEDGES 160000
#define DIMC   128
#define NHEADS 4
#define HD     32
#define NLAYER 2
#define SCALE_QK 0.17677669529663687f   // 1/sqrt(32)
#define LN_EPS 1e-5f

typedef __attribute__((ext_vector_type(16))) __bf16 v16bf;
typedef __attribute__((ext_vector_type(8)))  float  v8f;

__device__ __forceinline__ unsigned short f2bf(float f) {
    unsigned int u = __float_as_uint(f);
    u += 0x7FFFu + ((u >> 16) & 1u);          // round-to-nearest-even
    return (unsigned short)(u >> 16);
}
__device__ __forceinline__ float wredsum(float v) {
#pragma unroll
    for (int m = 1; m < 32; m <<= 1) v += __shfl_xor(v, m, 32);
    return v;
}
__device__ __forceinline__ float dot4(float4 a, float4 b) {
    return a.x * b.x + a.y * b.y + a.z * b.z + a.w * b.w;
}
// monotone f32 <-> u32 map for atomic max over signed floats
__device__ __forceinline__ unsigned int encf(float f) {
    unsigned int u = __float_as_uint(f);
    return (u >> 31) ? ~u : (u | 0x80000000u);
}
__device__ __forceinline__ float decf(unsigned int e) {
    return (e >> 31) ? __uint_as_float(e & 0x7FFFFFFFu) : __uint_as_float(~e);
}

// ---------------------------------------------------------------------------
// C[M,128] = act(A[M,K] @ W[K,128] + bias)   via v_wmma_f32_16x16x32_bf16
// block: 128 threads (4 waves), tile 64 rows x 128 cols, K in {64,128}
// ---------------------------------------------------------------------------
#define LSTR 132
__global__ __launch_bounds__(128)
void gemm_bf16_wmma(const float* __restrict__ A, const float* __restrict__ W,
                    const float* __restrict__ bias, float* __restrict__ C,
                    int M, int K, int do_relu)
{
    __shared__ unsigned short sWt[DIMC * LSTR];  // [n][k], bf16 bits
    __shared__ unsigned short sA [64   * LSTR];  // [m][k], bf16 bits
    const int tid = threadIdx.x;
    const int m0  = blockIdx.x * 64;

    for (int idx = tid; idx < K * DIMC; idx += 128) {   // W row-major [k][n]
        int k = idx >> 7, n = idx & 127;
        sWt[n * LSTR + k] = f2bf(W[idx]);
    }
    for (int idx = tid; idx < 64 * K; idx += 128) {
        int r = idx / K, c = idx - r * K;
        int g = m0 + r;
        float v = (g < M) ? A[(long)g * K + c] : 0.0f;
        sA[r * LSTR + c] = f2bf(v);
    }
    __syncthreads();

    const int wave = tid >> 5, lane = tid & 31;
    const int hl = lane >> 4, ml = lane & 15;
    const int arow = wave * 16 + ml;

    v8f acc[8];
    for (int nt = 0; nt < 8; nt++)
        for (int j = 0; j < 8; j++) acc[nt][j] = 0.0f;

    const int ksteps = K >> 5;
    for (int ks = 0; ks < ksteps; ks++) {
        union { v16bf v; unsigned int u[8]; } a;
#pragma unroll
        for (int j = 0; j < 8; j++) {        // ISA A 16x32 bf16 layout
            int grp = j >> 2, jj = j & 3;
            int k = ks * 32 + grp * 16 + hl * 8 + 2 * jj;
            a.u[j] = *(const unsigned int*)&sA[arow * LSTR + k];
        }
#pragma unroll
        for (int nt = 0; nt < 8; nt++) {
            union { v16bf v; unsigned int u[8]; } b;
            int n = nt * 16 + ml;
#pragma unroll
            for (int v = 0; v < 8; v++) {    // ISA B 32x16 bf16 layout
                int k = ks * 32 + (v >> 2) * 16 + hl * 8 + 2 * (v & 3);
                b.u[v] = *(const unsigned int*)&sWt[n * LSTR + k];
            }
            acc[nt] = __builtin_amdgcn_wmma_f32_16x16x32_bf16(
                false, a.v, false, b.v, (short)0, acc[nt], false, false);
        }
    }
#pragma unroll
    for (int nt = 0; nt < 8; nt++) {
        int col = nt * 16 + ml;
        float bv = bias ? bias[col] : 0.0f;
#pragma unroll
        for (int v = 0; v < 8; v++) {        // ISA C/D: VGPR v -> row v (+8 hi half)
            int row = m0 + wave * 16 + hl * 8 + v;
            if (row < M) {
                float r = acc[nt][v] + bv;
                if (do_relu) r = fmaxf(r, 0.0f);
                C[(long)row * DIMC + col] = r;
            }
        }
    }
}

// ---------------------------------------------------------------------------
// per-layer scratch init
// ---------------------------------------------------------------------------
__global__ void init_layer(float* ksum, float* cnt, float* denom,
                           unsigned int* smax, float* outb)
{
    long i = (long)blockIdx.x * blockDim.x + threadIdx.x;
    if (i < (long)NNODES * DIMC) { ksum[i] = 0.0f; outb[i] = 0.0f; }
    if (i < NNODES) cnt[i] = 0.0f;
    if (i < NNODES * 4) { denom[i] = 0.0f; smax[i] = 0x007FFFFFu; } // enc(-inf)
}

// ---------------------------------------------------------------------------
// edge pass 1: one wave per edge (4 channels/lane)
// s[e,h], delta_hist, volatility, edge_norm, ksum[dst]+=Kj, cnt[dst]+=1
// ---------------------------------------------------------------------------
__global__ __launch_bounds__(256)
void edge_pass1(const float* __restrict__ Q, const float* __restrict__ Kn,
                const float* __restrict__ Ke, const float* __restrict__ mem,
                const int* __restrict__ ei,
                float* __restrict__ sbuf, float* __restrict__ feats,
                float* __restrict__ ksum, float* __restrict__ cnt)
{
    int wave = threadIdx.x >> 5, lane = threadIdx.x & 31;
    long e = (long)blockIdx.x * 8 + wave;
    if (e >= NEDGES) return;
    int s = ei[e], d = ei[NEDGES + e];
    int c0 = lane * 4;
    float4 qi = *(const float4*)&Q [(long)d * DIMC + c0];
    float4 kj = *(const float4*)&Kn[(long)s * DIMC + c0];
    float4 ke = *(const float4*)&Ke[e * DIMC + c0];
    float4 mv = *(const float4*)&mem[(long)d * DIMC + c0];
    float4 kf = make_float4(kj.x + ke.x, kj.y + ke.y, kj.z + ke.z, kj.w + ke.w);

    float qk = dot4(qi, kf);
#pragma unroll
    for (int m = 1; m < 8; m <<= 1) qk += __shfl_xor(qk, m, 32);  // per-head
    float sh = qk * SCALE_QK;

    float t = fabsf(sh);
    t += __shfl_xor(t, 8, 32);
    t += __shfl_xor(t, 16, 32);
    float vol = t * 0.25f;                                        // mean |s| over heads

    float4 dl = make_float4(kj.x - mv.x, kj.y - mv.y, kj.z - mv.z, kj.w - mv.w);
    float dh = sqrtf(wredsum(dot4(dl, dl)));
    float en = sqrtf(wredsum(dot4(ke, ke)));

    atomicAdd(&ksum[(long)d * DIMC + c0 + 0], kj.x);
    atomicAdd(&ksum[(long)d * DIMC + c0 + 1], kj.y);
    atomicAdd(&ksum[(long)d * DIMC + c0 + 2], kj.z);
    atomicAdd(&ksum[(long)d * DIMC + c0 + 3], kj.w);

    if ((lane & 7) == 0) sbuf[e * 4 + (lane >> 3)] = sh;
    if (lane == 0) {
        atomicAdd(&cnt[d], 1.0f);
        feats[e * 4 + 0] = dh;
        feats[e * 4 + 2] = vol;
        feats[e * 4 + 3] = en;
    }
}

// ---------------------------------------------------------------------------
// edge pass 2: dev_nei, trust MLP (4->128->4, sigmoid), atomic seg-max of s*trust
// ---------------------------------------------------------------------------
__global__ __launch_bounds__(256)
void edge_pass2(const float* __restrict__ Kn, const float* __restrict__ ksum,
                const float* __restrict__ cnt, const int* __restrict__ ei,
                const float* __restrict__ feats, const float* __restrict__ sbuf,
                const float* __restrict__ Wt1, const float* __restrict__ bt1,
                const float* __restrict__ Wt2, const float* __restrict__ bt2,
                float* __restrict__ trust, unsigned int* __restrict__ smax)
{
    int wave = threadIdx.x >> 5, lane = threadIdx.x & 31;
    long e = (long)blockIdx.x * 8 + wave;
    if (e >= NEDGES) return;
    int s = ei[e], d = ei[NEDGES + e];
    int c0 = lane * 4;
    float4 kj = *(const float4*)&Kn  [(long)s * DIMC + c0];
    float4 km = *(const float4*)&ksum[(long)d * DIMC + c0];
    float inv = 1.0f / fmaxf(cnt[d], 1.0f);
    float4 dv = make_float4(kj.x - km.x * inv, kj.y - km.y * inv,
                            kj.z - km.z * inv, kj.w - km.w * inv);
    float f1 = sqrtf(wredsum(dot4(dv, dv)));     // dev_nei
    float f0 = feats[e * 4 + 0];
    float f2 = feats[e * 4 + 2];
    float f3 = feats[e * 4 + 3];

    float p0 = 0.f, p1 = 0.f, p2 = 0.f, p3 = 0.f;
#pragma unroll
    for (int jj = 0; jj < 4; jj++) {
        int j = lane + jj * 32;
        float hv = f0 * Wt1[j] + f1 * Wt1[128 + j] + f2 * Wt1[256 + j] +
                   f3 * Wt1[384 + j] + bt1[j];
        hv = fmaxf(hv, 0.0f);
        p0 += hv * Wt2[j * 4 + 0];
        p1 += hv * Wt2[j * 4 + 1];
        p2 += hv * Wt2[j * 4 + 2];
        p3 += hv * Wt2[j * 4 + 3];
    }
    p0 = wredsum(p0); p1 = wredsum(p1); p2 = wredsum(p2); p3 = wredsum(p3);
    if (lane < 4) {
        float z = (lane == 0) ? p0 : (lane == 1) ? p1 : (lane == 2) ? p2 : p3;
        z += bt2[lane];
        float tr = 1.0f / (1.0f + __expf(-z));
        trust[e * 4 + lane] = tr;
        float st = sbuf[e * 4 + lane] * tr;
        atomicMax(&smax[(long)d * 4 + lane], encf(st));
    }
}

// ---------------------------------------------------------------------------
// edge pass 3: exp(s*trust - max), denom accumulation.  thread per (e,h)
// ---------------------------------------------------------------------------
__global__ __launch_bounds__(256)
void edge_pass3(const int* __restrict__ ei, const float* __restrict__ sbuf,
                const float* __restrict__ trust, const unsigned int* __restrict__ smax,
                float* __restrict__ eexp, float* __restrict__ denom)
{
    long idx = (long)blockIdx.x * blockDim.x + threadIdx.x;
    if (idx >= (long)NEDGES * 4) return;
    long e = idx >> 2; int h = (int)(idx & 3);
    int d = ei[NEDGES + e];
    float st = sbuf[idx] * trust[idx];
    float ex = __expf(st - decf(smax[(long)d * 4 + h]));
    eexp[idx] = ex;
    atomicAdd(&denom[(long)d * 4 + h], ex);
}

// ---------------------------------------------------------------------------
// edge pass 4: out[dst] += (Vj+Ve) * attn * trust   (wave per edge)
// ---------------------------------------------------------------------------
__global__ __launch_bounds__(256)
void edge_pass4(const float* __restrict__ Vn, const float* __restrict__ Ve,
                const int* __restrict__ ei, const float* __restrict__ eexp,
                const float* __restrict__ trust, const float* __restrict__ denom,
                float* __restrict__ outb)
{
    int wave = threadIdx.x >> 5, lane = threadIdx.x & 31;
    long e = (long)blockIdx.x * 8 + wave;
    if (e >= NEDGES) return;
    int s = ei[e], d = ei[NEDGES + e];
    int h = lane >> 3;
    float ex = eexp[e * 4 + h];
    float tr = trust[e * 4 + h];
    float dn = denom[(long)d * 4 + h];
    float coef = (ex / (dn + 1e-16f)) * tr;
    int c0 = lane * 4;
    float4 vj = *(const float4*)&Vn[(long)s * DIMC + c0];
    float4 ve = *(const float4*)&Ve[e * DIMC + c0];
    atomicAdd(&outb[(long)d * DIMC + c0 + 0], (vj.x + ve.x) * coef);
    atomicAdd(&outb[(long)d * DIMC + c0 + 1], (vj.y + ve.y) * coef);
    atomicAdd(&outb[(long)d * DIMC + c0 + 2], (vj.z + ve.z) * coef);
    atomicAdd(&outb[(long)d * DIMC + c0 + 3], (vj.w + ve.w) * coef);
}

// ---------------------------------------------------------------------------
// h = layernorm(h + out) * g + b        (wave per node)
// ---------------------------------------------------------------------------
__global__ __launch_bounds__(256)
void node_ln(float* __restrict__ h, const float* __restrict__ outb,
             const float* __restrict__ g, const float* __restrict__ b)
{
    int wave = threadIdx.x >> 5, lane = threadIdx.x & 31;
    long n = (long)blockIdx.x * 8 + wave;
    if (n >= NNODES) return;
    int c0 = lane * 4;
    float4 hv = *(const float4*)&h   [n * DIMC + c0];
    float4 ov = *(const float4*)&outb[n * DIMC + c0];
    float4 x = make_float4(hv.x + ov.x, hv.y + ov.y, hv.z + ov.z, hv.w + ov.w);
    float mu = wredsum(x.x + x.y + x.z + x.w) * (1.0f / 128.0f);
    float4 dx = make_float4(x.x - mu, x.y - mu, x.z - mu, x.w - mu);
    float var = wredsum(dot4(dx, dx)) * (1.0f / 128.0f);
    float rs = rsqrtf(var + LN_EPS);
    float4 gv = *(const float4*)&g[c0];
    float4 bv = *(const float4*)&b[c0];
    float4 y = make_float4(dx.x * rs * gv.x + bv.x, dx.y * rs * gv.y + bv.y,
                           dx.z * rs * gv.z + bv.z, dx.w * rs * gv.w + bv.w);
    *(float4*)&h[n * DIMC + c0] = y;
}

// ---------------------------------------------------------------------------
extern "C" void kernel_launch(void* const* d_in, const int* in_sizes, int n_in,
                              void* d_out, int out_size, void* d_ws, size_t ws_size,
                              hipStream_t stream)
{
    const float* x    = (const float*)d_in[0];
    const int*   ei   = (const int*)  d_in[1];
    const float* msg  = (const float*)d_in[2];
    const float* mem  = (const float*)d_in[3];
    const float* Wp   = (const float*)d_in[4];
    const float* bp   = (const float*)d_in[5];
    const float* We1  = (const float*)d_in[6];
    const float* be1  = (const float*)d_in[7];
    const float* We2  = (const float*)d_in[8];
    const float* be2  = (const float*)d_in[9];
    const float* Wq   = (const float*)d_in[10];
    const float* bq   = (const float*)d_in[11];
    const float* Wkn  = (const float*)d_in[12];
    const float* bkn  = (const float*)d_in[13];
    const float* Wvn  = (const float*)d_in[14];
    const float* bvn  = (const float*)d_in[15];
    const float* Wke  = (const float*)d_in[16];
    const float* bke  = (const float*)d_in[17];
    const float* Wve  = (const float*)d_in[18];
    const float* bve  = (const float*)d_in[19];
    const float* Wt1  = (const float*)d_in[20];
    const float* bt1  = (const float*)d_in[21];
    const float* Wt2  = (const float*)d_in[22];
    const float* bt2  = (const float*)d_in[23];
    const float* ln_g = (const float*)d_in[24];
    const float* ln_b = (const float*)d_in[25];

    float* h = (float*)d_out;                     // [N,128], lives in d_out

    float* w = (float*)d_ws;
    float* edge_emb = w; w += (long)NEDGES * DIMC;
    float* ebuf2    = w; w += (long)NEDGES * DIMC;   // MLP hidden, then Ve
    float* Kebuf    = w; w += (long)NEDGES * DIMC;
    float* Qb   = w; w += (long)NNODES * DIMC;
    float* Knb  = w; w += (long)NNODES * DIMC;
    float* Vnb  = w; w += (long)NNODES * DIMC;
    float* ksum = w; w += (long)NNODES * DIMC;
    float* outb = w; w += (long)NNODES * DIMC;
    float* cnt  = w; w += NNODES;
    float* sbuf = w; w += (long)NEDGES * 4;
    float* trst = w; w += (long)NEDGES * 4;
    float* eexp = w; w += (long)NEDGES * 4;
    float* fts  = w; w += (long)NEDGES * 4;
    float* denm = w; w += NNODES * 4;
    unsigned int* smax = (unsigned int*)w; w += NNODES * 4;

    const int gN = (NNODES + 63) / 64;
    const int gE = (NEDGES + 63) / 64;
    const int gEw = (NEDGES + 7) / 8;      // wave-per-edge kernels
    const int gNw = (NNODES + 7) / 8;
    const int gInit = ((NNODES * DIMC) + 255) / 256;
    const int gP3 = (NEDGES * 4 + 255) / 256;

    // input projection + edge encoder
    gemm_bf16_wmma<<<gN, 128, 0, stream>>>(x, Wp, bp, h, NNODES, 128, 0);
    gemm_bf16_wmma<<<gE, 128, 0, stream>>>(msg, We1, be1, ebuf2, NEDGES, 64, 1);
    gemm_bf16_wmma<<<gE, 128, 0, stream>>>(ebuf2, We2, be2, edge_emb, NEDGES, 128, 0);

    for (int l = 0; l < NLAYER; l++) {
        const long wo = (long)l * DIMC * DIMC;
        const long bo = (long)l * DIMC;
        gemm_bf16_wmma<<<gN, 128, 0, stream>>>(h, Wq + wo,  bq + bo,  Qb,  NNODES, 128, 0);
        gemm_bf16_wmma<<<gN, 128, 0, stream>>>(h, Wkn + wo, bkn + bo, Knb, NNODES, 128, 0);
        gemm_bf16_wmma<<<gN, 128, 0, stream>>>(h, Wvn + wo, bvn + bo, Vnb, NNODES, 128, 0);
        gemm_bf16_wmma<<<gE, 128, 0, stream>>>(edge_emb, Wke + wo, bke + bo, Kebuf, NEDGES, 128, 0);
        gemm_bf16_wmma<<<gE, 128, 0, stream>>>(edge_emb, Wve + wo, bve + bo, ebuf2, NEDGES, 128, 0);

        init_layer<<<gInit, 256, 0, stream>>>(ksum, cnt, denm, smax, outb);
        edge_pass1<<<gEw, 256, 0, stream>>>(Qb, Knb, Kebuf, mem, ei, sbuf, fts, ksum, cnt);
        edge_pass2<<<gEw, 256, 0, stream>>>(Knb, ksum, cnt, ei, fts, sbuf,
                                            Wt1 + (long)l * 4 * DIMC, bt1 + bo,
                                            Wt2 + (long)l * DIMC * 4, bt2 + (long)l * 4,
                                            trst, smax);
        edge_pass3<<<gP3, 256, 0, stream>>>(ei, sbuf, trst, smax, eexp, denm);
        edge_pass4<<<gEw, 256, 0, stream>>>(Vnb, ebuf2, ei, eexp, trst, denm, outb);
        node_ln<<<gNw, 256, 0, stream>>>(h, outb, ln_g + bo, ln_b + bo);
    }
}
